// GlobalModelNodeAttention_24472723652621
// MI455X (gfx1250) — matrix operand order repaired and checked by reading, not compiled, not run on hardware
//
#include <hip/hip_runtime.h>

// ---------------------------------------------------------------------------
// CDNA5 (gfx1250) fused GraphNet global-model kernel.
// Dominant cost: per-node gating MLP  [131072,512]x[512,512] -> relu ->
// [131072,512]x[512,256]  (~103 GFLOP)  ==> v_wmma_f32_16x16x32_bf16
// (bf16 A/B, f32 accumulate). HBM traffic ~150-270MB (~10us @ 23.3TB/s) vs
// ~20-40us of matrix-core work => compute bound. Weights are pre-swizzled to
// the native B-fragment lane layout (contiguous 32B per lane); activations are
// staged in LDS as bf16 with a padded stride for conflict-free ds_load_b128
// A-fragment reads. GEMMs are register-blocked 2 N-tiles wide: two
// independent WMMA accumulator chains per wave + half the LDS A traffic.
// ---------------------------------------------------------------------------

typedef __attribute__((ext_vector_type(16))) __bf16 v16bf;
typedef __attribute__((ext_vector_type(8)))  __bf16 v8bf;
typedef __attribute__((ext_vector_type(8)))  float  v8f;

namespace {
constexpr int NNODES = 131072;
constexpr int NGRAPH = 512;
constexpr int FXC    = 256;   // F_X (== F_U == F_U_OUT)
constexpr int KDIM   = 512;   // F_X + F_U  (GEMM1 K)
constexpr int HIDC   = 512;   // hidden width (GEMM2 K)
constexpr int ROWS   = 64;    // node rows per block (4 waves x 16 rows)
constexpr int LDA    = 520;   // LDS row stride in bf16 (+8 pad -> 4-bank rotate)
constexpr int SMEM_BYTES = 2 * ROWS * LDA * (int)sizeof(__bf16); // A + H = 130KB
}

// Load a 16x32 bf16 A-fragment for this lane from an LDS row.
// ISA layout: lane<16 holds K {0..7, 16..23}, lane>=16 holds K {8..15, 24..31}
// (caller bakes the +8 half-lane offset into `base`). Two ds_load_b128.
__device__ __forceinline__ v16bf load_a_frag(const __bf16* base) {
    v8bf lo = *(const v8bf*)(base);
    v8bf hi = *(const v8bf*)(base + 16);
    v16bf a;
#pragma unroll
    for (int i = 0; i < 8; ++i) { a[i] = lo[i]; a[i + 8] = hi[i]; }
    return a;
}

// ---------------------------------------------------------------------------
// Weight pre-swizzle: f32 [K][Nw] row-major  ->  bf16 B-fragment layout:
//   out[((kc*NT + j)*32 + lane)*16 + t] = W[kc*32 + (lane>=16?8:0) + (t&7) + (t>=8?16:0)][j*16 + lane%16]
// so each lane's B fragment is one contiguous 32-byte load.
// ---------------------------------------------------------------------------
__global__ void swizzle_w_bf16(const float* __restrict__ W,
                               __bf16* __restrict__ out, int K, int Nw) {
    const int idx = blockIdx.x * blockDim.x + threadIdx.x;
    if (idx >= K * Nw) return;
    const int NT  = Nw >> 4;
    const int t   = idx & 15;
    const int ln  = (idx >> 4) & 31;
    const int blk = idx >> 9;
    const int j   = blk % NT;
    const int kc  = blk / NT;
    const int srcK = kc * 32 + ((ln >= 16) ? 8 : 0) + (t & 7) + ((t >= 8) ? 16 : 0);
    const int srcN = j * 16 + (ln & 15);
    out[idx] = (__bf16)W[(size_t)srcK * Nw + srcN];
}

__global__ void zero_f32(float* __restrict__ p, int n) {
    const int i = blockIdx.x * blockDim.x + threadIdx.x;
    if (i < n) p[i] = 0.0f;
}

__global__ void count_nodes(const int* __restrict__ batch,
                            float* __restrict__ counts, int n) {
    const int i = blockIdx.x * blockDim.x + threadIdx.x;
    if (i < n) unsafeAtomicAdd(&counts[batch[i]], 1.0f);
}

// ---------------------------------------------------------------------------
// Fused node kernel: A = concat(x, u[batch]) -> relu(A@Wg1+bg1)@Wg2+bg2 = a
//                    sums[batch] += x * a   (global f32 atomics, mean later)
// 128 threads = 4 waves; each wave owns a 16-row subtile end-to-end.
// ---------------------------------------------------------------------------
__global__ __launch_bounds__(128) void node_attn_mlp(
    const float* __restrict__ x, const float* __restrict__ u,
    const int* __restrict__ batch,
    const __bf16* __restrict__ W1, const float* __restrict__ b1,
    const __bf16* __restrict__ W2, const float* __restrict__ b2,
    float* __restrict__ sums) {
    extern __shared__ __bf16 smem[];
    __bf16* Ash = smem;                 // [ROWS][LDA] bf16 concat(x,u)
    __bf16* Hsh = smem + ROWS * LDA;    // [ROWS][LDA] bf16 hidden

    const int tid    = threadIdx.x;
    const int lane   = tid & 31;
    const int wave   = tid >> 5;
    const int laneN  = lane & 15;
    const int hiHalf = (lane >> 4) & 1;   // 0: lanes 0-15, 1: lanes 16-31
    const int rowBase = blockIdx.x * ROWS;

    // ---- Stage concat(x, u[batch]) tile into LDS, f32 -> bf16 ----
    for (int i = tid; i < ROWS * (FXC / 4); i += 128) {
        const int r = i >> 6, c4 = i & 63;
        const float4 v = ((const float4*)(x + (size_t)(rowBase + r) * FXC))[c4];
        __bf16* d = Ash + r * LDA + c4 * 4;
        d[0] = (__bf16)v.x; d[1] = (__bf16)v.y; d[2] = (__bf16)v.z; d[3] = (__bf16)v.w;
    }
    for (int i = tid; i < ROWS * (FXC / 4); i += 128) {
        const int r = i >> 6, c4 = i & 63;
        const int g = batch[rowBase + r];
        const float4 v = ((const float4*)(u + (size_t)g * FXC))[c4];
        __bf16* d = Ash + r * LDA + FXC + c4 * 4;
        d[0] = (__bf16)v.x; d[1] = (__bf16)v.y; d[2] = (__bf16)v.z; d[3] = (__bf16)v.w;
    }
    __syncthreads();

    const int wrow = wave * 16;
    const __bf16* aRow = Ash + (wrow + laneN) * LDA + hiHalf * 8;

    // ---- GEMM1: [16 x 512] @ [512 x 512], +bias, relu -> Hsh (bf16) ----
    // 2-way N-tile register blocking: one A fragment feeds two independent
    // WMMA accumulator chains.
    for (int j = 0; j < HIDC / 16; j += 2) {
        const float bias0 = b1[j * 16 + laneN];
        const float bias1 = b1[(j + 1) * 16 + laneN];
        v8f acc0, acc1;
#pragma unroll
        for (int r = 0; r < 8; ++r) { acc0[r] = bias0; acc1[r] = bias1; }
#pragma unroll
        for (int kc = 0; kc < KDIM / 32; ++kc) {
            const v16bf av = load_a_frag(aRow + kc * 32);
            const __bf16* bp = W1 + (((size_t)kc * (HIDC / 16) + j) * 32 + lane) * 16;
            if (kc + 1 < KDIM / 32)
                __builtin_prefetch(bp + (HIDC / 16) * 512, 0, 3);  // WGP-scope
            const v16bf bv0 = *(const v16bf*)bp;
            const v16bf bv1 = *(const v16bf*)(bp + 512);  // j+1 tile: next 1KB block
            acc0 = __builtin_amdgcn_wmma_f32_16x16x32_bf16(
                false, av, false, bv0, (short)0, acc0, false, false);
            acc1 = __builtin_amdgcn_wmma_f32_16x16x32_bf16(
                false, av, false, bv1, (short)0, acc1, false, false);
        }
#pragma unroll
        for (int r = 0; r < 8; ++r) {   // C layout: M = hiHalf*8 + r, N = laneN
            const float h0 = acc0[r] > 0.0f ? acc0[r] : 0.0f;
            const float h1 = acc1[r] > 0.0f ? acc1[r] : 0.0f;
            __bf16* hd = Hsh + (wrow + hiHalf * 8 + r) * LDA + j * 16 + laneN;
            hd[0]  = (__bf16)h0;
            hd[16] = (__bf16)h1;
        }
    }
    // Each wave reads only its own H rows; per-wave DS ordering suffices.

    const __bf16* hRow = Hsh + (wrow + laneN) * LDA + hiHalf * 8;

    // ---- GEMM2: [16 x 512] @ [512 x 256], +bias; gate by x; scatter-add ----
    for (int j = 0; j < FXC / 16; j += 2) {
        const float bias0 = b2[j * 16 + laneN];
        const float bias1 = b2[(j + 1) * 16 + laneN];
        v8f acc0, acc1;
#pragma unroll
        for (int r = 0; r < 8; ++r) { acc0[r] = bias0; acc1[r] = bias1; }
#pragma unroll
        for (int kc = 0; kc < HIDC / 32; ++kc) {
            const v16bf av = load_a_frag(hRow + kc * 32);
            const __bf16* bp = W2 + (((size_t)kc * (FXC / 16) + j) * 32 + lane) * 16;
            if (kc + 1 < HIDC / 32)
                __builtin_prefetch(bp + (FXC / 16) * 512, 0, 3);   // WGP-scope
            const v16bf bv0 = *(const v16bf*)bp;
            const v16bf bv1 = *(const v16bf*)(bp + 512);
            acc0 = __builtin_amdgcn_wmma_f32_16x16x32_bf16(
                false, av, false, bv0, (short)0, acc0, false, false);
            acc1 = __builtin_amdgcn_wmma_f32_16x16x32_bf16(
                false, av, false, bv1, (short)0, acc1, false, false);
        }
#pragma unroll
        for (int r = 0; r < 8; ++r) {
            const int grow = rowBase + wrow + hiHalf * 8 + r;
            const int col  = j * 16 + laneN;
            const float xv0 = x[(size_t)grow * FXC + col];        // f32 gate (L2 hit)
            const float xv1 = x[(size_t)grow * FXC + col + 16];
            const int   g   = batch[grow];
            unsafeAtomicAdd(sums + (size_t)g * FXC + col,      xv0 * acc0[r]);
            unsafeAtomicAdd(sums + (size_t)g * FXC + col + 16, xv1 * acc1[r]);
        }
    }
}

// ---------------------------------------------------------------------------
// Final graph MLP: A = concat(sums/counts, u) [512,512] -> relu(.@Wp1+bp1)@Wp2+bp2
// Same WMMA tiling; 8 blocks x 64 graph rows.
// ---------------------------------------------------------------------------
__global__ __launch_bounds__(128) void graph_mlp(
    const float* __restrict__ sums, const float* __restrict__ counts,
    const float* __restrict__ u,
    const __bf16* __restrict__ W1, const float* __restrict__ b1,
    const __bf16* __restrict__ W2, const float* __restrict__ b2,
    float* __restrict__ out) {
    extern __shared__ __bf16 smem[];
    __bf16* Ash = smem;
    __bf16* Hsh = smem + ROWS * LDA;

    const int tid    = threadIdx.x;
    const int lane   = tid & 31;
    const int wave   = tid >> 5;
    const int laneN  = lane & 15;
    const int hiHalf = (lane >> 4) & 1;
    const int rowBase = blockIdx.x * ROWS;

    for (int i = tid; i < ROWS * (FXC / 4); i += 128) {
        const int r = i >> 6, c4 = i & 63;
        const float inv = 1.0f / fmaxf(counts[rowBase + r], 1.0f);
        const float4 v = ((const float4*)(sums + (size_t)(rowBase + r) * FXC))[c4];
        __bf16* d = Ash + r * LDA + c4 * 4;
        d[0] = (__bf16)(v.x * inv); d[1] = (__bf16)(v.y * inv);
        d[2] = (__bf16)(v.z * inv); d[3] = (__bf16)(v.w * inv);
    }
    for (int i = tid; i < ROWS * (FXC / 4); i += 128) {
        const int r = i >> 6, c4 = i & 63;
        const float4 v = ((const float4*)(u + (size_t)(rowBase + r) * FXC))[c4];
        __bf16* d = Ash + r * LDA + FXC + c4 * 4;
        d[0] = (__bf16)v.x; d[1] = (__bf16)v.y; d[2] = (__bf16)v.z; d[3] = (__bf16)v.w;
    }
    __syncthreads();

    const int wrow = wave * 16;
    const __bf16* aRow = Ash + (wrow + laneN) * LDA + hiHalf * 8;

    for (int j = 0; j < HIDC / 16; j += 2) {
        const float bias0 = b1[j * 16 + laneN];
        const float bias1 = b1[(j + 1) * 16 + laneN];
        v8f acc0, acc1;
#pragma unroll
        for (int r = 0; r < 8; ++r) { acc0[r] = bias0; acc1[r] = bias1; }
#pragma unroll
        for (int kc = 0; kc < KDIM / 32; ++kc) {
            const v16bf av = load_a_frag(aRow + kc * 32);
            const __bf16* bp = W1 + (((size_t)kc * (HIDC / 16) + j) * 32 + lane) * 16;
            const v16bf bv0 = *(const v16bf*)bp;
            const v16bf bv1 = *(const v16bf*)(bp + 512);
            acc0 = __builtin_amdgcn_wmma_f32_16x16x32_bf16(
                false, av, false, bv0, (short)0, acc0, false, false);
            acc1 = __builtin_amdgcn_wmma_f32_16x16x32_bf16(
                false, av, false, bv1, (short)0, acc1, false, false);
        }
#pragma unroll
        for (int r = 0; r < 8; ++r) {
            const float h0 = acc0[r] > 0.0f ? acc0[r] : 0.0f;
            const float h1 = acc1[r] > 0.0f ? acc1[r] : 0.0f;
            __bf16* hd = Hsh + (wrow + hiHalf * 8 + r) * LDA + j * 16 + laneN;
            hd[0]  = (__bf16)h0;
            hd[16] = (__bf16)h1;
        }
    }

    const __bf16* hRow = Hsh + (wrow + laneN) * LDA + hiHalf * 8;

    for (int j = 0; j < FXC / 16; j += 2) {
        const float bias0 = b2[j * 16 + laneN];
        const float bias1 = b2[(j + 1) * 16 + laneN];
        v8f acc0, acc1;
#pragma unroll
        for (int r = 0; r < 8; ++r) { acc0[r] = bias0; acc1[r] = bias1; }
#pragma unroll
        for (int kc = 0; kc < HIDC / 32; ++kc) {
            const v16bf av = load_a_frag(hRow + kc * 32);
            const __bf16* bp = W2 + (((size_t)kc * (FXC / 16) + j) * 32 + lane) * 16;
            const v16bf bv0 = *(const v16bf*)bp;
            const v16bf bv1 = *(const v16bf*)(bp + 512);
            acc0 = __builtin_amdgcn_wmma_f32_16x16x32_bf16(
                false, av, false, bv0, (short)0, acc0, false, false);
            acc1 = __builtin_amdgcn_wmma_f32_16x16x32_bf16(
                false, av, false, bv1, (short)0, acc1, false, false);
        }
#pragma unroll
        for (int r = 0; r < 8; ++r) {
            const int grow = rowBase + wrow + hiHalf * 8 + r;
            const int col  = j * 16 + laneN;
            out[(size_t)grow * FXC + col]      = acc0[r];
            out[(size_t)grow * FXC + col + 16] = acc1[r];
        }
    }
}

// ---------------------------------------------------------------------------
extern "C" void kernel_launch(void* const* d_in, const int* in_sizes, int n_in,
                              void* d_out, int out_size, void* d_ws, size_t ws_size,
                              hipStream_t stream) {
    (void)in_sizes; (void)n_in; (void)out_size; (void)ws_size;
    const float* x    = (const float*)d_in[0];
    // d_in[1] edge_index, d_in[2] edge_attr: unused by the global model
    const float* u    = (const float*)d_in[3];
    const int*   batch = (const int*)d_in[4];
    const float* Wg1  = (const float*)d_in[5];
    const float* bg1  = (const float*)d_in[6];
    const float* Wg2  = (const float*)d_in[7];
    const float* bg2  = (const float*)d_in[8];
    const float* Wp1  = (const float*)d_in[9];
    const float* bp1  = (const float*)d_in[10];
    const float* Wp2  = (const float*)d_in[11];
    const float* bp2  = (const float*)d_in[12];
    float* out = (float*)d_out;

    // Workspace layout (bytes):
    //   [0, 524288)         sums   512*256 f32
    //   [524288, 526336)    counts 512 f32
    //   [526336, 1050624)   Wg1 swizzled bf16 (512x512)
    //   [1050624, 1312768)  Wg2 swizzled bf16 (512x256)
    //   [1312768, 1837056)  Wp1 swizzled bf16 (512x512)
    //   [1837056, 2099200)  Wp2 swizzled bf16 (512x256)
    char* ws = (char*)d_ws;
    float*  sums   = (float*)(ws);
    float*  counts = (float*)(ws + 524288);
    __bf16* W1s    = (__bf16*)(ws + 526336);
    __bf16* W2s    = (__bf16*)(ws + 1050624);
    __bf16* P1s    = (__bf16*)(ws + 1312768);
    __bf16* P2s    = (__bf16*)(ws + 1837056);

    // Re-zero accumulators every launch (graph replays must be self-contained).
    zero_f32<<<(131584 + 255) / 256, 256, 0, stream>>>(sums, 131584);

    swizzle_w_bf16<<<(512 * 512 + 255) / 256, 256, 0, stream>>>(Wg1, W1s, 512, 512);
    swizzle_w_bf16<<<(512 * 256 + 255) / 256, 256, 0, stream>>>(Wg2, W2s, 512, 256);
    swizzle_w_bf16<<<(512 * 512 + 255) / 256, 256, 0, stream>>>(Wp1, P1s, 512, 512);
    swizzle_w_bf16<<<(512 * 256 + 255) / 256, 256, 0, stream>>>(Wp2, P2s, 512, 256);

    count_nodes<<<NNODES / 256, 256, 0, stream>>>(batch, counts, NNODES);

    node_attn_mlp<<<NNODES / ROWS, 128, SMEM_BYTES, stream>>>(
        x, u, batch, W1s, bg1, W2s, bg2, sums);

    graph_mlp<<<NGRAPH / ROWS, 128, SMEM_BYTES, stream>>>(
        sums, counts, u, P1s, bp1, P2s, bp2, out);
}
